// AdaptiveGNN_77575699300448
// MI455X (gfx1250) — matrix operand order repaired and checked by reading, not compiled, not run on hardware
//
#include <hip/hip_runtime.h>

typedef __attribute__((ext_vector_type(16))) __bf16 v16bf;
typedef __attribute__((ext_vector_type(8)))  float  v8f;

#define GEMM_ROWS 128

static inline int cdiv_i(long a, long b) { return (int)((a + b - 1) / b); }

// ---------------------------------------------------------------- utilities
__global__ void __launch_bounds__(256) k_fill(float* __restrict__ p, float v, long n) {
  long i = (long)blockIdx.x * blockDim.x + threadIdx.x;
  long G = (long)gridDim.x * blockDim.x;
  for (; i < n; i += G) p[i] = v;
}

__global__ void __launch_bounds__(256) k_degree(const int* __restrict__ idx,
                                                float* __restrict__ deg, int n) {
  int i = blockIdx.x * blockDim.x + threadIdx.x;
  int G = gridDim.x * blockDim.x;
  for (; i < n; i += G) atomicAdd(&deg[idx[i]], 1.0f);
}

__global__ void __launch_bounds__(256) k_invsqrt(float* __restrict__ d, int n) {
  int i = blockIdx.x * blockDim.x + threadIdx.x;
  int G = gridDim.x * blockDim.x;
  for (; i < n; i += G) {
    float v = d[i];
    d[i] = (v > 0.f) ? rsqrtf(fmaxf(v, 1.f)) : 0.f;
  }
}

// ------------------------------------------------------- WMMA bf16 GEMM
// C[N,64] = A[N,K] @ W[K,64] (+ bias), fp32 in/out, bf16 WMMA f32-accum.
// KPAD in {32,64} compile-time; K <= KPAD. Block = 256 threads (8 waves),
// 128 rows/block. W staged transposed in LDS so B fragments are contiguous
// ds_load_b128 pairs; B fragments are wave-invariant (tn fixed per wave) and
// loaded once. Full-tile blocks take an unguarded store path.
template <int KPAD, bool BIAS>
__global__ void __launch_bounds__(256)
k_gemm_wmma(const float* __restrict__ A, const float* __restrict__ W,
            const float* __restrict__ bias, float* __restrict__ C, int N, int K) {
  constexpr int LDP  = KPAD + 8;            // row stride (16B-aligned rows)
  constexpr int SH   = (KPAD == 64) ? 6 : 5;
  constexpr int MSK  = KPAD - 1;
  constexpr int KTIL = KPAD / 32;
  __shared__ __bf16 sA[GEMM_ROWS][LDP];     // A tile, row-major [row][k]
  __shared__ __bf16 sBt[64][LDP];           // W transposed: sBt[col][k]

  const int tid = threadIdx.x;

  // stage transposed weights (coalesced global read, one-time LDS write)
  for (int idx = tid; idx < KPAD * 64; idx += 256) {
    int r = idx >> 6, c = idx & 63;         // r = k, c = col
    sBt[c][r] = (r < K) ? (__bf16)W[r * 64 + c] : (__bf16)0.f;
  }
  // stage A tile (fp32 -> bf16), zero-pad cols >= K and rows >= N
  const int row0 = blockIdx.x * GEMM_ROWS;
  for (int idx = tid; idx < GEMM_ROWS * KPAD; idx += 256) {
    int r = idx >> SH, c = idx & MSK;
    int g = row0 + r;
    sA[r][c] = (g < N && c < K) ? (__bf16)A[(size_t)g * K + c] : (__bf16)0.f;
  }
  __syncthreads();

  const int wid  = tid >> 5;
  const int lane = tid & 31;
  const int half = lane >> 4;   // 0: lanes 0-15, 1: lanes 16-31
  const int l15  = lane & 15;

  // tile column fixed per wave -> B fragments loaded once, reused 4x
  const int tn  = (wid & 3) * 16;
  const int col = tn + l15;
  const float bv = BIAS ? bias[col] : 0.f;

  v16bf bf[KTIL];
#pragma unroll
  for (int kk = 0; kk < KTIL; ++kk) {
    // lanes 0-15: K=k0..k0+15; lanes 16-31: K=k0+16..k0+31 (contiguous)
    const int kc = kk * 32 + half * 16;
#pragma unroll
    for (int i = 0; i < 16; ++i) bf[kk][i] = sBt[col][kc + i];
  }

  const bool full = (row0 + GEMM_ROWS) <= N;  // uniform across block

#pragma unroll
  for (int rt = 0; rt < 4; ++rt) {
    const int tm = (wid >> 2) * 16 + rt * 32;
    v8f acc = {};
#pragma unroll
    for (int kk = 0; kk < KTIL; ++kk) {
      // A fragment: lanes 0-15 (M=l15): K=k0..k0+15;
      //             lanes 16-31:        K=k0+16..k0+31 (contiguous)
      v16bf a;
      const int m  = tm + l15;
      const int kb = kk * 32 + half * 16;
#pragma unroll
      for (int i = 0; i < 8; ++i) {
        a[i]     = sA[m][kb + i];
        a[i + 8] = sA[m][kb + 8 + i];
      }
      acc = __builtin_amdgcn_wmma_f32_16x16x32_bf16(
          /*neg_a=*/false, a, /*neg_b=*/false, bf[kk],
          /*c_mod=*/(short)0, acc, /*reuse_a=*/false, /*reuse_b=*/false);
    }
    // C layout: element j -> (M = tm + half*8 + j, N = col)
    const int rbase = row0 + tm + half * 8;
    float* cp = C + (size_t)rbase * 64 + col;
    if (full) {
#pragma unroll
      for (int j = 0; j < 8; ++j) cp[(size_t)j * 64] = acc[j] + bv;
    } else {
#pragma unroll
      for (int j = 0; j < 8; ++j)
        if (rbase + j < N) cp[(size_t)j * 64] = acc[j] + bv;
    }
  }
}

// ------------------------------------------------------ edge scatter (GCN)
// one wave per edge; each lane handles a float2 slice of the 64-wide row.
__global__ void __launch_bounds__(256)
k_scatter(const float* __restrict__ h, const int* __restrict__ src,
          const int* __restrict__ dst, const float* __restrict__ inv_s,
          const float* __restrict__ inv_d, float* __restrict__ acc, int nE) {
  const int lane = threadIdx.x & 31;
  int wave = (int)((blockIdx.x * blockDim.x + threadIdx.x) >> 5);
  const int nW = (int)((gridDim.x * blockDim.x) >> 5);
  for (int e = wave; e < nE; e += nW) {
    const int s = src[e], d = dst[e];
    const float norm = inv_s[s] * inv_d[d];
    const float2 v = ((const float2*)(h + (size_t)s * 64))[lane];
    float* dp = acc + (size_t)d * 64 + 2 * lane;
    atomicAdd(dp,     v.x * norm);
    atomicAdd(dp + 1, v.y * norm);
  }
}

// self-loop contribution for the zone-zone relation (deterministic add)
__global__ void __launch_bounds__(256)
k_self_loop(const float* __restrict__ h, const float* __restrict__ inv_s,
            const float* __restrict__ inv_d, float* __restrict__ acc, long n) {
  long i = (long)blockIdx.x * blockDim.x + threadIdx.x;
  long G = (long)gridDim.x * blockDim.x;
  for (; i < n; i += G) {
    long node = i >> 6;
    acc[i] += h[i] * inv_s[node] * inv_d[node];
  }
}

// zone finalize: mean of 3 relations, biases folded in, optional relu
__global__ void __launch_bounds__(256)
k_finalize_zone(float* __restrict__ x, const float* __restrict__ b1,
                const float* __restrict__ b2, const float* __restrict__ b3,
                long n, int relu) {
  long i = (long)blockIdx.x * blockDim.x + threadIdx.x;
  long G = (long)gridDim.x * blockDim.x;
  for (; i < n; i += G) {
    int f = (int)(i & 63);
    float v = (x[i] + b1[f] + b2[f] + b3[f]) * (1.f / 3.f);
    x[i] = relu ? fmaxf(v, 0.f) : v;
  }
}

__global__ void __launch_bounds__(256)
k_finalize(float* __restrict__ x, const float* __restrict__ b, long n, int relu) {
  long i = (long)blockIdx.x * blockDim.x + threadIdx.x;
  long G = (long)gridDim.x * blockDim.x;
  for (; i < n; i += G) {
    int f = (int)(i & 63);
    float v = x[i] + b[f];
    x[i] = relu ? fmaxf(v, 0.f) : v;
  }
}

// column-wise mean pooling into pooled[64] (grid*block multiple of 64)
__global__ void __launch_bounds__(256)
k_mean_pool(const float* __restrict__ x, float* __restrict__ pooled,
            long n, float scale) {
  long g = (long)blockIdx.x * blockDim.x + threadIdx.x;
  long G = (long)gridDim.x * blockDim.x;
  int f = (int)(g & 63);
  float s = 0.f;
  for (long i = g; i < n; i += G) s += x[i];
  atomicAdd(&pooled[f], s * scale);
}

// tiny MLP head: comb[256] -> fc1(relu, 64) -> fc2 -> scalar
__global__ void __launch_bounds__(256)
k_head(const float* __restrict__ pooled, const float* __restrict__ bt_emb,
       const int* __restrict__ bt_idx, const float* __restrict__ W1,
       const float* __restrict__ b1, const float* __restrict__ W2,
       const float* __restrict__ b2, float* __restrict__ out) {
  __shared__ float comb[256];
  __shared__ float hbuf[64];
  int t = threadIdx.x;
  comb[t] = (t < 192) ? pooled[t] : bt_emb[(size_t)(*bt_idx) * 64 + (t - 192)];
  __syncthreads();
  if (t < 64) {
    float s = b1[t];
#pragma unroll 8
    for (int k = 0; k < 256; ++k) s += comb[k] * W1[k * 64 + t];
    hbuf[t] = fmaxf(s, 0.f);
  }
  __syncthreads();
  if (t == 0) {
    float s = b2[0];
    for (int k = 0; k < 64; ++k) s += hbuf[k] * W2[k];
    out[0] = s;
  }
}

// ---------------------------------------------------------------- launcher
extern "C" void kernel_launch(void* const* d_in, const int* in_sizes, int n_in,
                              void* d_out, int out_size, void* d_ws, size_t ws_size,
                              hipStream_t stream) {
  (void)n_in; (void)out_size; (void)ws_size;
  const float* x_zone  = (const float*)d_in[0];
  const float* x_equip = (const float*)d_in[1];
  const float* x_surf  = (const float*)d_in[2];
  const int* e_cc_s = (const int*)d_in[3];
  const int* e_cc_d = (const int*)d_in[4];
  const int* e_ct_s = (const int*)d_in[5];
  const int* e_ct_d = (const int*)d_in[6];
  const int* e_bt_s = (const int*)d_in[7];
  const int* e_bt_d = (const int*)d_in[8];
  const int* e_bo_s = (const int*)d_in[9];
  const int* e_bo_d = (const int*)d_in[10];
  const int* e_bb_s = (const int*)d_in[11];
  const int* e_bb_d = (const int*)d_in[12];
  const int* bt_idx = (const int*)d_in[13];

  int p = 14;
  const float* embzW = (const float*)d_in[p++]; const float* embzB = (const float*)d_in[p++];
  const float* embeW = (const float*)d_in[p++]; const float* embeB = (const float*)d_in[p++];
  const float* embsW = (const float*)d_in[p++]; const float* embsB = (const float*)d_in[p++];
  // RELS order: 0=connects 1=contains 2=belongs_to 3=bounds 4=bounded_by
  const float *c1W[5], *c1B[5], *c2W[5], *c2B[5];
  for (int r = 0; r < 5; ++r) { c1W[r] = (const float*)d_in[p++]; c1B[r] = (const float*)d_in[p++]; }
  for (int r = 0; r < 5; ++r) { c2W[r] = (const float*)d_in[p++]; c2B[r] = (const float*)d_in[p++]; }
  const float* bt_emb = (const float*)d_in[p++];
  const float* fc1W = (const float*)d_in[p++]; const float* fc1B = (const float*)d_in[p++];
  const float* fc2W = (const float*)d_in[p++]; const float* fc2B = (const float*)d_in[p++];

  const int NZ = in_sizes[0] / 32, NE = in_sizes[1] / 16, NS = in_sizes[2] / 24;
  const int nCC = in_sizes[3], nCT = in_sizes[5], nBT = in_sizes[7];
  const int nBO = in_sizes[9], nBB = in_sizes[11];
  const long FZ = (long)NZ * 64, FE = (long)NE * 64, FS = (long)NS * 64;

  // --- workspace carve (float units, 256B-aligned chunks)
  float* ws = (float*)d_ws;
  size_t off = 0;
  auto carve = [&](long n) -> float* {
    float* q = ws + off;
    off += (size_t)((n + 63) & ~63LL);
    return q;
  };
  float *F0z = carve(FZ), *F0e = carve(FE), *F0s = carve(FS);
  float *F1z = carve(FZ), *F1e = carve(FE), *F1s = carve(FS);
  float *Hb  = carve(FS);  // shared h = x@W staging, sized for largest type
  float *dg_cc_s = carve(NZ), *dg_cc_d = carve(NZ);
  float *dg_ct_s = carve(NZ), *dg_ct_d = carve(NE);
  float *dg_bt_s = carve(NE), *dg_bt_d = carve(NZ);
  float *dg_bo_s = carve(NS), *dg_bo_d = carve(NZ);
  float *dg_bb_s = carve(NZ), *dg_bb_d = carve(NS);
  float *pooled  = carve(192);

  auto gridc = [](long n) { int b = cdiv_i(n, 256); return b > 8192 ? 8192 : b; };
  auto fill = [&](float* q, float v, long n) {
    k_fill<<<gridc(n), 256, 0, stream>>>(q, v, n);
  };
  auto degree = [&](const int* idx, float* dg, int n) {
    k_degree<<<gridc(n), 256, 0, stream>>>(idx, dg, n);
  };
  auto invs = [&](float* dg, int n) {
    k_invsqrt<<<gridc(n), 256, 0, stream>>>(dg, n);
  };
  auto gemm = [&](const float* A, const float* W, const float* B, float* C, int N, int K) {
    int blocks = cdiv_i(N, GEMM_ROWS);
    if (K > 32) {
      if (B) k_gemm_wmma<64, true ><<<blocks, 256, 0, stream>>>(A, W, B, C, N, K);
      else   k_gemm_wmma<64, false><<<blocks, 256, 0, stream>>>(A, W, B, C, N, K);
    } else {
      if (B) k_gemm_wmma<32, true ><<<blocks, 256, 0, stream>>>(A, W, B, C, N, K);
      else   k_gemm_wmma<32, false><<<blocks, 256, 0, stream>>>(A, W, B, C, N, K);
    }
  };
  auto scatter = [&](const float* h, const int* s, const int* d, const float* is,
                     const float* id, float* acc, int nE) {
    int blocks = cdiv_i(nE, 8); if (blocks > 262144) blocks = 262144;
    k_scatter<<<blocks, 256, 0, stream>>>(h, s, d, is, id, acc, nE);
  };

  // --- degrees (computed once; self-loops on connects => init 1.0)
  fill(dg_cc_s, 1.f, NZ); fill(dg_cc_d, 1.f, NZ);
  fill(dg_ct_s, 0.f, NZ); fill(dg_ct_d, 0.f, NE);
  fill(dg_bt_s, 0.f, NE); fill(dg_bt_d, 0.f, NZ);
  fill(dg_bo_s, 0.f, NS); fill(dg_bo_d, 0.f, NZ);
  fill(dg_bb_s, 0.f, NZ); fill(dg_bb_d, 0.f, NS);
  degree(e_cc_s, dg_cc_s, nCC); degree(e_cc_d, dg_cc_d, nCC);
  degree(e_ct_s, dg_ct_s, nCT); degree(e_ct_d, dg_ct_d, nCT);
  degree(e_bt_s, dg_bt_s, nBT); degree(e_bt_d, dg_bt_d, nBT);
  degree(e_bo_s, dg_bo_s, nBO); degree(e_bo_d, dg_bo_d, nBO);
  degree(e_bb_s, dg_bb_s, nBB); degree(e_bb_d, dg_bb_d, nBB);
  invs(dg_cc_s, NZ); invs(dg_cc_d, NZ);
  invs(dg_ct_s, NZ); invs(dg_ct_d, NE);
  invs(dg_bt_s, NE); invs(dg_bt_d, NZ);
  invs(dg_bo_s, NS); invs(dg_bo_d, NZ);
  invs(dg_bb_s, NZ); invs(dg_bb_d, NS);

  // --- input embeddings (WMMA GEMMs, K zero-padded to 32)
  gemm(x_zone,  embzW, embzB, F0z, NZ, 32);
  gemm(x_equip, embeW, embeB, F0e, NE, 16);
  gemm(x_surf,  embsW, embsB, F0s, NS, 24);

  // --- one hetero-conv layer
  auto conv = [&](const float* iz, const float* ie, const float* isf,
                  float* oz, float* oe, float* os,
                  const float* const* Wr, const float* const* Br, int relu) {
    fill(oz, 0.f, FZ); fill(oe, 0.f, FE); fill(os, 0.f, FS);
    // zone <- connects (zone, self-loops) + belongs_to (equip) + bounds (surf)
    gemm(iz, Wr[0], nullptr, Hb, NZ, 64);
    k_self_loop<<<gridc(FZ), 256, 0, stream>>>(Hb, dg_cc_s, dg_cc_d, oz, FZ);
    scatter(Hb, e_cc_s, e_cc_d, dg_cc_s, dg_cc_d, oz, nCC);
    gemm(ie, Wr[2], nullptr, Hb, NE, 64);
    scatter(Hb, e_bt_s, e_bt_d, dg_bt_s, dg_bt_d, oz, nBT);
    gemm(isf, Wr[3], nullptr, Hb, NS, 64);
    scatter(Hb, e_bo_s, e_bo_d, dg_bo_s, dg_bo_d, oz, nBO);
    k_finalize_zone<<<gridc(FZ), 256, 0, stream>>>(oz, Br[0], Br[2], Br[3], FZ, relu);
    // equipment <- contains (zone)
    gemm(iz, Wr[1], nullptr, Hb, NZ, 64);
    scatter(Hb, e_ct_s, e_ct_d, dg_ct_s, dg_ct_d, oe, nCT);
    k_finalize<<<gridc(FE), 256, 0, stream>>>(oe, Br[1], FE, relu);
    // surface <- bounded_by (zone)
    gemm(iz, Wr[4], nullptr, Hb, NZ, 64);
    scatter(Hb, e_bb_s, e_bb_d, dg_bb_s, dg_bb_d, os, nBB);
    k_finalize<<<gridc(FS), 256, 0, stream>>>(os, Br[4], FS, relu);
  };

  conv(F0z, F0e, F0s, F1z, F1e, F1s, c1W, c1B, /*relu=*/1);
  conv(F1z, F1e, F1s, F0z, F0e, F0s, c2W, c2B, /*relu=*/0);

  // --- pooling + head
  fill(pooled, 0.f, 192);
  k_mean_pool<<<512, 256, 0, stream>>>(F0z, pooled + 0,   FZ, 1.f / (float)NZ);
  k_mean_pool<<<512, 256, 0, stream>>>(F0e, pooled + 64,  FE, 1.f / (float)NE);
  k_mean_pool<<<512, 256, 0, stream>>>(F0s, pooled + 128, FS, 1.f / (float)NS);
  k_head<<<1, 256, 0, stream>>>(pooled, bt_emb, bt_idx, fc1W, fc1B, fc2W, fc2B,
                                (float*)d_out);
}
